// InGramRelationLayer_64046552318128
// MI455X (gfx1250) — matrix operand order repaired and checked by reading, not compiled, not run on hardware
//
#include <hip/hip_runtime.h>
#include <math.h>

#define NEG_SLOPE 0.2f
#define D_IN 256
#define H    8
#define DH   32
#define NCAT 768   // [Ph | Pt(+attn_b) | A(+aggr_b)] columns

typedef __attribute__((ext_vector_type(16))) _Float16 v16h;
typedef __attribute__((ext_vector_type(8)))  float    v8f;

union Frag16 { v16h v; unsigned int u[8]; };
union FragC  { v8f  v; float f[8]; };

// ---------------- prep: fp32 -> f16 hi/lo split ----------------
__global__ void split_to_f16x2(const float* __restrict__ src,
                               _Float16* __restrict__ hi,
                               _Float16* __restrict__ lo, size_t n) {
  size_t i = (size_t)blockIdx.x * blockDim.x + threadIdx.x;
  if (i >= n) return;
  float x = src[i];
  _Float16 h = (_Float16)x;
  hi[i] = h;
  lo[i] = (_Float16)(x - (float)h);
}

// Build combined weight Wcat (768x256) = [attn_w[:, :256]; attn_w[:, 256:]; aggr_w],
// split to f16 hi/lo, and build combined bias [0; attn_b; aggr_b].
__global__ void prep_weights(const float* __restrict__ attn_w,
                             const float* __restrict__ aggr_w,
                             const float* __restrict__ attn_b,
                             const float* __restrict__ aggr_b,
                             _Float16* __restrict__ wh,
                             _Float16* __restrict__ wl,
                             float* __restrict__ biascat) {
  int idx = blockIdx.x * blockDim.x + threadIdx.x;
  if (idx >= NCAT * D_IN) return;
  int j = idx >> 8, k = idx & 255;
  float w;
  if (j < 256)      w = attn_w[j * 512 + k];
  else if (j < 512) w = attn_w[(j - 256) * 512 + 256 + k];
  else              w = aggr_w[(j - 512) * 256 + k];
  _Float16 h = (_Float16)w;
  wh[idx] = h;
  wl[idx] = (_Float16)(w - (float)h);
  if (k == 0)
    biascat[j] = (j < 256) ? 0.0f : (j < 512 ? attn_b[j - 256] : aggr_b[j - 512]);
}

__global__ void init_buffers(float* __restrict__ out, float* __restrict__ seg_max,
                             float* __restrict__ seg_sum, int N) {
  int i = blockIdx.x * blockDim.x + threadIdx.x;
  if (i < N * 256) out[i] = 0.0f;
  if (i < N * H) { seg_max[i] = -INFINITY; seg_sum[i] = 0.0f; }
}

// ---------------- node-level GEMM: P = emb(Nx256) @ Wcat.T -> (N x 768) -------
// f16x3 split-precision accumulation via v_wmma_f32_16x16x32_f16.
// One wave computes one 16x16 output tile.
__global__ void __launch_bounds__(128)
gemm_f16x3_wmma(const _Float16* __restrict__ eh, const _Float16* __restrict__ el,
                const _Float16* __restrict__ wh, const _Float16* __restrict__ wl,
                const float* __restrict__ biascat, float* __restrict__ P, int N) {
  const int TILES_N = NCAT / 16;  // 48
  int wave = (int)((blockIdx.x * blockDim.x + threadIdx.x) >> 5);
  int lane = threadIdx.x & 31;
  int tile_m = wave / TILES_N;
  int tile_n = wave - tile_m * TILES_N;
  int tiles_m = (N + 15) >> 4;
  if (tile_m >= tiles_m) return;  // wave-uniform: EXEC stays all-ones for WMMA

  int mrow  = tile_m * 16 + (lane & 15);
  int mload = mrow < N ? mrow : N - 1;      // clamp loads; stores guarded below
  int khA   = (lane >> 4) * 8;              // A: K-half 0-7 / 8-15 per lane group
  int ncol  = tile_n * 16 + (lane & 15);
  int kbB   = (lane >> 4) * 8;              // B: K-half 0-7 / 8-15 per lane group

  const _Float16* arh = eh + (size_t)mload * D_IN;
  const _Float16* arl = el + (size_t)mload * D_IN;
  const _Float16* brh = wh + (size_t)ncol * D_IN;  // row n of Wcat == column n of B
  const _Float16* brl = wl + (size_t)ncol * D_IN;

  v8f acc = {};
  #pragma unroll
  for (int k0 = 0; k0 < D_IN; k0 += 32) {
    Frag16 ah, al, bh, bl;
    #pragma unroll
    for (int i = 0; i < 4; ++i) {
      ah.u[i]     = *(const unsigned int*)(arh + k0 + khA + 2 * i);
      ah.u[4 + i] = *(const unsigned int*)(arh + k0 + 16 + khA + 2 * i);
      al.u[i]     = *(const unsigned int*)(arl + k0 + khA + 2 * i);
      al.u[4 + i] = *(const unsigned int*)(arl + k0 + 16 + khA + 2 * i);
    }
    #pragma unroll
    for (int i = 0; i < 4; ++i) {
      bh.u[i]     = *(const unsigned int*)(brh + k0 + kbB + 2 * i);
      bh.u[4 + i] = *(const unsigned int*)(brh + k0 + 16 + kbB + 2 * i);
      bl.u[i]     = *(const unsigned int*)(brl + k0 + kbB + 2 * i);
      bl.u[4 + i] = *(const unsigned int*)(brl + k0 + 16 + kbB + 2 * i);
    }
    acc = __builtin_amdgcn_wmma_f32_16x16x32_f16(false, ah.v, false, bh.v, (short)0, acc, false, false);
    acc = __builtin_amdgcn_wmma_f32_16x16x32_f16(false, al.v, false, bh.v, (short)0, acc, false, false);
    acc = __builtin_amdgcn_wmma_f32_16x16x32_f16(false, ah.v, false, bl.v, (short)0, acc, false, false);
  }

  FragC c; c.v = acc;
  float bias = biascat[ncol];
  #pragma unroll
  for (int r = 0; r < 8; ++r) {
    int M = tile_m * 16 + r + ((lane >> 4) << 3);   // C layout: M = r + 8*(lane>=16)
    if (M < N) P[(size_t)M * NCAT + ncol] = c.f[r] + bias;
  }
}

// ---------------- per-edge attention logits + segment max ----------------
__device__ inline void atomicMaxF(float* addr, float val) {
  if (val >= 0.0f) atomicMax((int*)addr, __float_as_int(val));
  else             atomicMin((unsigned int*)addr, __float_as_uint(val));
}

__global__ void __launch_bounds__(256)
edge_attn(const float* __restrict__ P, const int* __restrict__ head,
          const int* __restrict__ tail, const int* __restrict__ bins,
          const float* __restrict__ attn_vec, const float* __restrict__ attn_bin,
          float* __restrict__ attn_rv, float* __restrict__ seg_max, int E) {
  int edge = blockIdx.x * 8 + (threadIdx.x >> 5);
  if (edge >= E) return;
  int lane = threadIdx.x & 31;
  int hn = head[edge], tn = tail[edge];
  const float* ph = P + (size_t)hn * NCAT;        // Ph row
  const float* pt = P + (size_t)tn * NCAT + 256;  // Pt row (bias folded)
  float s[H];
  #pragma unroll
  for (int j = 0; j < H; ++j) {
    int i = j * DH + lane;                        // head j, dh = lane
    float v = ph[i] + pt[i];
    v = (v >= 0.0f) ? v : NEG_SLOPE * v;          // LeakyReLU
    s[j] = v * attn_vec[i];
  }
  #pragma unroll
  for (int m = 16; m >= 1; m >>= 1) {
    #pragma unroll
    for (int j = 0; j < H; ++j) s[j] += __shfl_xor(s[j], m, 32);
  }
  if (lane == 0) {
    int b = bins[edge];
    #pragma unroll
    for (int j = 0; j < H; ++j) {
      float r = s[j] + attn_bin[b * H + j];
      attn_rv[(size_t)edge * H + j] = r;
      atomicMaxF(&seg_max[hn * H + j], r);
    }
  }
}

// ---------------- exp(x - segmax) + segment sum ----------------
__global__ void edge_exp(const int* __restrict__ head, float* __restrict__ attn_rv,
                         const float* __restrict__ seg_max, float* __restrict__ seg_sum,
                         int E) {
  int idx = blockIdx.x * blockDim.x + threadIdx.x;
  if (idx >= E * H) return;
  int e = idx >> 3, j = idx & 7;
  int hn = head[e];
  float v = expf(attn_rv[idx] - seg_max[hn * H + j]);
  attn_rv[idx] = v;                       // overwrite raw logit with exp value
  atomicAdd(&seg_sum[hn * H + j], v);
}

// ---------------- normalize + weighted scatter-add into output -------------
__global__ void __launch_bounds__(256)
edge_aggr(const float* __restrict__ P, const int* __restrict__ head,
          const int* __restrict__ tail, const float* __restrict__ attn_rv,
          const float* __restrict__ seg_sum, float* __restrict__ out, int E) {
  int edge = blockIdx.x * 8 + (threadIdx.x >> 5);
  if (edge >= E) return;
  int lane = threadIdx.x & 31;
  int hn = head[edge], tn = tail[edge];
  const float* a = P + (size_t)tn * NCAT + 512;   // aggr row (bias folded)
  float* o = out + (size_t)hn * 256;
  #pragma unroll
  for (int j = 0; j < H; ++j) {
    float beta = attn_rv[(size_t)edge * H + j] / (seg_sum[hn * H + j] + 1e-16f);
    atomicAdd(&o[j * DH + lane], beta * a[j * DH + lane]);
  }
}

// ---------------- host launcher ----------------
extern "C" void kernel_launch(void* const* d_in, const int* in_sizes, int n_in,
                              void* d_out, int out_size, void* d_ws, size_t ws_size,
                              hipStream_t stream) {
  const float* emb      = (const float*)d_in[0];
  const int*   head     = (const int*)d_in[1];
  const int*   tail     = (const int*)d_in[2];
  const int*   bins     = (const int*)d_in[3];
  const float* attn_w   = (const float*)d_in[4];
  const float* attn_b   = (const float*)d_in[5];
  const float* attn_bin = (const float*)d_in[6];
  const float* attn_vec = (const float*)d_in[7];
  const float* aggr_w   = (const float*)d_in[8];
  const float* aggr_b   = (const float*)d_in[9];
  int N = in_sizes[0] / D_IN;
  int E = in_sizes[1];
  float* out = (float*)d_out;

  // carve workspace (256B-aligned slices)
  char* ws = (char*)d_ws;
  size_t off = 0;
  auto take = [&](size_t bytes) -> char* {
    char* p = ws + off;
    off += (bytes + 255) & ~(size_t)255;
    return p;
  };
  float*    P       = (float*)   take((size_t)N * NCAT * sizeof(float));
  _Float16* emb_h   = (_Float16*)take((size_t)N * D_IN * sizeof(_Float16));
  _Float16* emb_l   = (_Float16*)take((size_t)N * D_IN * sizeof(_Float16));
  _Float16* w_h     = (_Float16*)take((size_t)NCAT * D_IN * sizeof(_Float16));
  _Float16* w_l     = (_Float16*)take((size_t)NCAT * D_IN * sizeof(_Float16));
  float*    biascat = (float*)   take((size_t)NCAT * sizeof(float));
  float*    attn_rv = (float*)   take((size_t)E * H * sizeof(float));
  float*    seg_max = (float*)   take((size_t)N * H * sizeof(float));
  float*    seg_sum = (float*)   take((size_t)N * H * sizeof(float));

  size_t nemb = (size_t)N * D_IN;
  split_to_f16x2<<<(int)((nemb + 255) / 256), 256, 0, stream>>>(emb, emb_h, emb_l, nemb);
  prep_weights<<<(NCAT * D_IN + 255) / 256, 256, 0, stream>>>(
      attn_w, aggr_w, attn_b, aggr_b, w_h, w_l, biascat);
  init_buffers<<<(N * 256 + 255) / 256, 256, 0, stream>>>(out, seg_max, seg_sum, N);

  int tiles_m = (N + 15) >> 4;
  long long waves = (long long)tiles_m * (NCAT / 16);
  int gemm_blocks = (int)((waves * 32 + 127) / 128);
  gemm_f16x3_wmma<<<gemm_blocks, 128, 0, stream>>>(emb_h, emb_l, w_h, w_l, biascat, P, N);

  edge_attn<<<(E + 7) / 8, 256, 0, stream>>>(P, head, tail, bins, attn_vec, attn_bin,
                                             attn_rv, seg_max, E);
  edge_exp<<<(E * H + 255) / 256, 256, 0, stream>>>(head, attn_rv, seg_max, seg_sum, E);
  edge_aggr<<<(E + 7) / 8, 256, 0, stream>>>(P, head, tail, attn_rv, seg_sum, out, E);
}
